// Guided_Attn_66245575573783
// MI455X (gfx1250) — matrix operand order repaired and checked by reading, not compiled, not run on hardware
//
#include <hip/hip_runtime.h>

// ---------------------------------------------------------------------------
// Guided attention for MI455X (gfx1250, wave32, WMMA f16 16x16x32).
// Pipeline:
//   K0a pack_data:    x, mask  -> channels-last f16 (xh, maskh)
//   K0b pack_weights: mq_w1/mk_w1 -> per-tap f16 [2][9][64][64]; ca_w1/2 -> f16
//   K1  conv3_wmma:   t = relu(conv3x3(mask)) both paths, WMMA implicit GEMM
//   K2  qkv_prep:     pq/pk (f16 [B][N][32], d padded 8->32) and pv (f16 [B][64][N])
//                     (bilinear 1/2 downsample == 2x2 avg pool)
//   K3  flash_attn:   fused QK^T + online softmax + PV, 1 wave = 16 queries
//   K4a upsample:     bilinear 2x upsample of attention output -> f16 chan-last
//   K4b final_conv:   1x1 -> relu -> 1x1 (WMMA) + residual, store f32 NCHW
// ---------------------------------------------------------------------------

typedef __attribute__((ext_vector_type(16))) _Float16 v16h;
typedef __attribute__((ext_vector_type(8)))  _Float16 v8h;
typedef __attribute__((ext_vector_type(8)))  float    v8f;

union VU { v16h v; v8h h[2]; };

__device__ __forceinline__ v8f vzero8() {
    v8f z;
#pragma unroll
    for (int i = 0; i < 8; ++i) z[i] = 0.f;
    return z;
}

__device__ __forceinline__ v16h hzero16() {
    v16h z;
#pragma unroll
    for (int i = 0; i < 16; ++i) z[i] = (_Float16)0.f;
    return z;
}

__device__ __forceinline__ v8f wmma_f16(v16h a, v16h b, v8f c) {
    // D = A(16x32 f16) x B(32x16 f16) + C(16x16 f32)
    return __builtin_amdgcn_wmma_f32_16x16x32_f16(false, a, false, b,
                                                  (short)0, c, false, false);
}

// reduce across the 16 lanes of each wave half (xor masks 1,2,4,8 stay in-half)
__device__ __forceinline__ float hmax16(float v) {
#pragma unroll
    for (int m = 1; m <= 8; m <<= 1) v = fmaxf(v, __shfl_xor(v, m, 32));
    return v;
}
__device__ __forceinline__ float hsum16(float v) {
#pragma unroll
    for (int m = 1; m <= 8; m <<= 1) v += __shfl_xor(v, m, 32);
    return v;
}

#define HW  16384   // 128*128
#define NN  4096    // 64*64

// ---------------------------------------------------------------------------
// K0a: transpose NCHW f32 -> [b][pixel][64] f16 for x and mask
__global__ void pack_data(const float* __restrict__ x, const float* __restrict__ mask,
                          _Float16* __restrict__ xh, _Float16* __restrict__ mh) {
    int t = blockIdx.x * blockDim.x + threadIdx.x;   // 65536 = B*HW
    int b = t >> 14, p = t & (HW - 1);
    const float* xs = x    + (size_t)b * 64 * HW + p;
    const float* ms = mask + (size_t)b * 64 * HW + p;
    _Float16* xd = xh + (size_t)t * 64;
    _Float16* md = mh + (size_t)t * 64;
#pragma unroll
    for (int c = 0; c < 64; ++c) {
        xd[c] = (_Float16)xs[(size_t)c * HW];
        md[c] = (_Float16)ms[(size_t)c * HW];
    }
}

// K0b: weights -> f16. wtap layout [path][tap][o][i]
__global__ void pack_weights(const float* __restrict__ mqw1, const float* __restrict__ mkw1,
                             const float* __restrict__ caw1, const float* __restrict__ caw2,
                             _Float16* __restrict__ wtap, _Float16* __restrict__ cw1h,
                             _Float16* __restrict__ cw2h) {
    int t = blockIdx.x * blockDim.x + threadIdx.x;
    if (t < 73728) {                       // 2*9*64*64
        int path = t / 36864; int r = t % 36864;
        int tap = r >> 12; int oi = r & 4095;
        int o = oi >> 6, i = oi & 63;
        const float* w = path ? mkw1 : mqw1;
        int ty = tap / 3, tx = tap % 3;
        wtap[t] = (_Float16)w[((o * 64 + i) * 3 + ty) * 3 + tx];
    } else if (t < 73728 + 4096) {
        int i = t - 73728;          cw1h[i] = (_Float16)caw1[i];
    } else if (t < 73728 + 8192) {
        int i = t - 73728 - 4096;   cw2h[i] = (_Float16)caw2[i];
    }
}

// ---------------------------------------------------------------------------
// K1: t[path] = relu(conv3x3(mask)) via WMMA implicit GEMM.
// 1 wave = 16 pixels of one image row; 9 taps x 2 K-steps x 4 M-tiles WMMAs.
__global__ __launch_bounds__(256) void conv3_wmma(
        const _Float16* __restrict__ mh, const _Float16* __restrict__ wtap,
        const float* __restrict__ bq1, const float* __restrict__ bk1,
        _Float16* __restrict__ th) {
    int wave = (int)((blockIdx.x * blockDim.x + threadIdx.x) >> 5); // 8192 tiles
    int lane = threadIdx.x & 31;
    int path = wave >> 12; int r = wave & 4095;
    int b = r >> 10; int r2 = r & 1023;
    int y = r2 >> 3; int x0 = (r2 & 7) << 4;
    int n = lane & 15, lhalf = lane >> 4;

    v8f acc[4];
#pragma unroll
    for (int mt = 0; mt < 4; ++mt) acc[mt] = vzero8();

    const _Float16* wbase = wtap + (size_t)path * 36864;
#pragma unroll
    for (int tap = 0; tap < 9; ++tap) {
        int dy = tap / 3 - 1, dx = tap % 3 - 1;
        int yy = y + dy, xx = x0 + n + dx;
        bool valid = (yy >= 0) && (yy < 128) && (xx >= 0) && (xx < 128);
        const _Float16* prow = mh + ((size_t)(b * HW + yy * 128 + xx)) * 64;
#pragma unroll
        for (int kc = 0; kc < 2; ++kc) {
            v16h bt;
            if (valid) {
                const _Float16* pr = prow + kc * 32 + lhalf * 16;
                VU u; u.h[0] = *(const v8h*)pr; u.h[1] = *(const v8h*)(pr + 8);
                bt = u.v;
            } else {
                bt = hzero16();
            }
#pragma unroll
            for (int mt = 0; mt < 4; ++mt) {
                const _Float16* ar = wbase + (size_t)tap * 4096
                                   + (mt * 16 + n) * 64 + kc * 32 + lhalf * 16;
                VU ua; ua.h[0] = *(const v8h*)ar; ua.h[1] = *(const v8h*)(ar + 8);
                acc[mt] = wmma_f16(ua.v, bt, acc[mt]);
            }
        }
    }
    const float* bias = path ? bk1 : bq1;
    _Float16* orow = th + ((size_t)((path * 4 + b) * HW + y * 128 + x0 + n)) * 64;
#pragma unroll
    for (int mt = 0; mt < 4; ++mt) {
        v8h st;
#pragma unroll
        for (int v = 0; v < 8; ++v) {
            int c = mt * 16 + lhalf * 8 + v;
            st[v] = (_Float16)fmaxf(acc[mt][v] + bias[c], 0.f);
        }
        *(v8h*)(orow + mt * 16 + lhalf * 8) = st;
    }
}

// ---------------------------------------------------------------------------
// K2: pq/pk (blend + 2x2 avgpool, f16 [B][N][32] zero-padded) and pv.
__global__ __launch_bounds__(256) void qkv_prep(
        const _Float16* __restrict__ xh, const _Float16* __restrict__ th,
        const float* __restrict__ Wq, const float* __restrict__ bq,
        const float* __restrict__ Wk, const float* __restrict__ bk,
        const float* __restrict__ Wv, const float* __restrict__ bv,
        const float* __restrict__ mqw2, const float* __restrict__ mqb2,
        const float* __restrict__ mkw2, const float* __restrict__ mkb2,
        const float* __restrict__ alpha1, const float* __restrict__ alpha2,
        _Float16* __restrict__ pq, _Float16* __restrict__ pk,
        _Float16* __restrict__ pv) {
    int t = blockIdx.x * blockDim.x + threadIdx.x;   // 16384 = B*N
    int b = t >> 12, n = t & (NN - 1);
    int hy = n >> 6, hx = n & 63;
    float a1 = alpha1[0], a2 = alpha2[0];

    float q[8], k[8];
#pragma unroll
    for (int d = 0; d < 8; ++d) { q[d] = 0.f; k[d] = 0.f; }
    float xavg[64];
#pragma unroll
    for (int c = 0; c < 64; ++c) xavg[c] = 0.f;

    for (int iy = 0; iy < 2; ++iy)
    for (int ix = 0; ix < 2; ++ix) {
        int p = (2 * hy + iy) * 128 + (2 * hx + ix);
        const _Float16* xr = xh + ((size_t)(b * HW + p)) * 64;
        const _Float16* t0 = th + ((size_t)(b * HW + p)) * 64;          // mq path
        const _Float16* t1 = th + ((size_t)((4 + b) * HW + p)) * 64;    // mk path
        float xq[8], xk[8], mq[8], mk[8];
#pragma unroll
        for (int d = 0; d < 8; ++d) {
            xq[d] = bq[d]; xk[d] = bk[d]; mq[d] = mqb2[d]; mk[d] = mkb2[d];
        }
#pragma unroll
        for (int c = 0; c < 64; ++c) {
            float xc  = (float)xr[c];
            float t0c = (float)t0[c], t1c = (float)t1[c];
            xavg[c] += 0.25f * xc;
#pragma unroll
            for (int d = 0; d < 8; ++d) {
                xq[d] += Wq[d * 64 + c] * xc;
                xk[d] += Wk[d * 64 + c] * xc;
                mq[d] += mqw2[d * 64 + c] * t0c;
                mk[d] += mkw2[d * 64 + c] * t1c;
            }
        }
#pragma unroll
        for (int d = 0; d < 8; ++d) {
            q[d] += 0.25f * (mq[d] * xq[d] * a1 + xq[d] * (1.f - a1));
            k[d] += 0.25f * (mk[d] * xk[d] * a2 + xk[d] * (1.f - a2));
        }
    }
    _Float16* pqr = pq + (size_t)t * 32;
    _Float16* pkr = pk + (size_t)t * 32;
#pragma unroll
    for (int d = 0; d < 8; ++d) { pqr[d] = (_Float16)q[d]; pkr[d] = (_Float16)k[d]; }
#pragma unroll
    for (int d = 8; d < 32; ++d) { pqr[d] = (_Float16)0.f; pkr[d] = (_Float16)0.f; }

    for (int o = 0; o < 64; ++o) {
        float s = bv[o];
#pragma unroll
        for (int c = 0; c < 64; ++c) s += Wv[o * 64 + c] * xavg[c];
        pv[((size_t)(b * 64 + o)) * NN + n] = (_Float16)s;
    }
}

// ---------------------------------------------------------------------------
// K3: fused flash attention. 1 wave = 16 queries; loop keys in blocks of 32.
__global__ __launch_bounds__(128) void flash_attn(
        const _Float16* __restrict__ pq, const _Float16* __restrict__ pk,
        const _Float16* __restrict__ pv, float* __restrict__ ao) {
    __shared__ __align__(16) _Float16 pls[4][16 * 32];
    int w = threadIdx.x >> 5, lane = threadIdx.x & 31;
    int n = lane & 15, lhalf = lane >> 4;
    int b = blockIdx.y;
    int mblk = blockIdx.x * 64 + w * 16;

    // Q tile in A layout (rows = queries, K = 32 padded features)
    const _Float16* qrow = pq + ((size_t)(b * NN + mblk + n)) * 32 + lhalf * 16;
    VU qa; qa.h[0] = *(const v8h*)qrow; qa.h[1] = *(const v8h*)(qrow + 8);

    float m_i[8], l_i[8];
#pragma unroll
    for (int v = 0; v < 8; ++v) { m_i[v] = -1e30f; l_i[v] = 0.f; }
    v8f oacc[4];
#pragma unroll
    for (int ct = 0; ct < 4; ++ct) oacc[ct] = vzero8();

    _Float16* myp = pls[w];
    const _Float16* pkb = pk + (size_t)b * NN * 32;
    const _Float16* pvb = pv + (size_t)b * 64 * NN;

    for (int k0 = 0; k0 < NN; k0 += 32) {
        if (k0 + 32 < NN) {   // global_prefetch_b8 for the next K block
            __builtin_prefetch(pkb + (size_t)(k0 + 32 + n) * 32, 0, 1);
            __builtin_prefetch(pvb + (size_t)n * NN + k0 + 32, 0, 1);
        }
        // S = Q x K^T for 32 keys (two B tiles)
        const _Float16* kr1 = pkb + (size_t)(k0 + n) * 32 + lhalf * 16;
        const _Float16* kr2 = kr1 + 16 * 32;
        VU u1, u2;
        u1.h[0] = *(const v8h*)kr1; u1.h[1] = *(const v8h*)(kr1 + 8);
        u2.h[0] = *(const v8h*)kr2; u2.h[1] = *(const v8h*)(kr2 + 8);
        v8f s1 = wmma_f16(qa.v, u1.v, vzero8());
        v8f s2 = wmma_f16(qa.v, u2.v, vzero8());

        // online softmax update (row m = v + 8*lhalf, col = n)
        float p1[8], p2[8], sc[8];
#pragma unroll
        for (int v = 0; v < 8; ++v) {
            float mv = hmax16(fmaxf(s1[v], s2[v]));
            float nm = fmaxf(m_i[v], mv);
            sc[v] = __expf(m_i[v] - nm);
            p1[v] = __expf(s1[v] - nm);
            p2[v] = __expf(s2[v] - nm);
            float rs = hsum16(p1[v] + p2[v]);
            l_i[v] = l_i[v] * sc[v] + rs;
            m_i[v] = nm;
        }
#pragma unroll
        for (int ct = 0; ct < 4; ++ct)
#pragma unroll
            for (int v = 0; v < 8; ++v) oacc[ct][v] *= sc[v];

        // transpose P (C/D layout -> A layout) through LDS
#pragma unroll
        for (int v = 0; v < 8; ++v) {
            int row = v + lhalf * 8;
            myp[row * 32 + n]      = (_Float16)p1[v];
            myp[row * 32 + 16 + n] = (_Float16)p2[v];
        }
        asm volatile("s_wait_dscnt 0" ::: "memory");
        const _Float16* par = myp + n * 32 + lhalf * 16;
        VU pa; pa.h[0] = *(const v8h*)par; pa.h[1] = *(const v8h*)(par + 8);

        // O += P x V  (B tile: 32 keys x 16 channels)
#pragma unroll
        for (int ct = 0; ct < 4; ++ct) {
            const _Float16* vr = pvb + (size_t)(ct * 16 + n) * NN + k0 + lhalf * 16;
            VU uv; uv.h[0] = *(const v8h*)vr; uv.h[1] = *(const v8h*)(vr + 8);
            oacc[ct] = wmma_f16(pa.v, uv.v, oacc[ct]);
        }
    }
    // normalize and store [b][query][64] f32
#pragma unroll
    for (int ct = 0; ct < 4; ++ct)
#pragma unroll
        for (int v = 0; v < 8; ++v) {
            int m = mblk + v + lhalf * 8;
            int c = ct * 16 + n;
            ao[((size_t)(b * NN + m)) * 64 + c] = oacc[ct][v] / l_i[v];
        }
}

// ---------------------------------------------------------------------------
// K4a: bilinear 2x upsample (align_corners=False) -> f16 channels-last
__global__ void upsample_bilin(const float* __restrict__ ao, _Float16* __restrict__ uh) {
    int t = blockIdx.x * blockDim.x + threadIdx.x;   // 65536
    int b = t >> 14, p = t & (HW - 1);
    int y = p >> 7, x = p & 127;
    float sy = y * 0.5f - 0.25f, sx = x * 0.5f - 0.25f;
    float fy = floorf(sy), fx = floorf(sx);
    float wy = sy - fy, wx = sx - fx;
    int y0 = min(63, max(0, (int)fy)), y1 = min(63, max(0, (int)fy + 1));
    int x0 = min(63, max(0, (int)fx)), x1 = min(63, max(0, (int)fx + 1));
    const float* r00 = ao + ((size_t)(b * NN + y0 * 64 + x0)) * 64;
    const float* r01 = ao + ((size_t)(b * NN + y0 * 64 + x1)) * 64;
    const float* r10 = ao + ((size_t)(b * NN + y1 * 64 + x0)) * 64;
    const float* r11 = ao + ((size_t)(b * NN + y1 * 64 + x1)) * 64;
    float w00 = (1.f - wy) * (1.f - wx), w01 = (1.f - wy) * wx;
    float w10 = wy * (1.f - wx),         w11 = wy * wx;
    _Float16* o = uh + (size_t)t * 64;
#pragma unroll
    for (int c = 0; c < 64; ++c)
        o[c] = (_Float16)(w00 * r00[c] + w01 * r01[c] + w10 * r10[c] + w11 * r11[c]);
}

// K4b: 1x1 -> relu -> 1x1 (WMMA, LDS between layers) + residual, f32 NCHW out
__global__ __launch_bounds__(256) void final_conv(
        const _Float16* __restrict__ uh, const _Float16* __restrict__ w1,
        const _Float16* __restrict__ w2, const float* __restrict__ b1,
        const float* __restrict__ b2, const float* __restrict__ x,
        float* __restrict__ out) {
    __shared__ __align__(16) _Float16 tls[8][16 * 64];
    int wv = threadIdx.x >> 5, lane = threadIdx.x & 31;
    int n = lane & 15, lhalf = lane >> 4;
    int tile = blockIdx.x * 8 + wv;           // 4096 tiles
    int b = tile >> 10; int p0 = (tile & 1023) << 4;

    const _Float16* ub = uh + ((size_t)(b * HW + p0 + n)) * 64;
    v8f a1[4];
#pragma unroll
    for (int mt = 0; mt < 4; ++mt) a1[mt] = vzero8();
#pragma unroll
    for (int kc = 0; kc < 2; ++kc) {
        const _Float16* br = ub + kc * 32 + lhalf * 16;
        VU u; u.h[0] = *(const v8h*)br; u.h[1] = *(const v8h*)(br + 8);
#pragma unroll
        for (int mt = 0; mt < 4; ++mt) {
            const _Float16* ar = w1 + (size_t)(mt * 16 + n) * 64 + kc * 32 + lhalf * 16;
            VU ua; ua.h[0] = *(const v8h*)ar; ua.h[1] = *(const v8h*)(ar + 8);
            a1[mt] = wmma_f16(ua.v, u.v, a1[mt]);
        }
    }
    _Float16* tl = tls[wv];
#pragma unroll
    for (int mt = 0; mt < 4; ++mt) {
        v8h st;
#pragma unroll
        for (int v = 0; v < 8; ++v) {
            int c = mt * 16 + lhalf * 8 + v;
            st[v] = (_Float16)fmaxf(a1[mt][v] + b1[c], 0.f);
        }
        *(v8h*)(tl + n * 64 + mt * 16 + lhalf * 8) = st;
    }
    asm volatile("s_wait_dscnt 0" ::: "memory");

    v8f a2[4];
#pragma unroll
    for (int mt = 0; mt < 4; ++mt) a2[mt] = vzero8();
#pragma unroll
    for (int kc = 0; kc < 2; ++kc) {
        const _Float16* br = tl + n * 64 + kc * 32 + lhalf * 16;
        VU u; u.h[0] = *(const v8h*)br; u.h[1] = *(const v8h*)(br + 8);
#pragma unroll
        for (int mt = 0; mt < 4; ++mt) {
            const _Float16* ar = w2 + (size_t)(mt * 16 + n) * 64 + kc * 32 + lhalf * 16;
            VU ua; ua.h[0] = *(const v8h*)ar; ua.h[1] = *(const v8h*)(ar + 8);
            a2[mt] = wmma_f16(ua.v, u.v, a2[mt]);
        }
    }
#pragma unroll
    for (int mt = 0; mt < 4; ++mt)
#pragma unroll
        for (int v = 0; v < 8; ++v) {
            int c = mt * 16 + lhalf * 8 + v;
            size_t idx = ((size_t)(b * 64 + c)) * HW + p0 + n;
            out[idx] = a2[mt][v] + b2[c] + x[idx];
        }
}

// ---------------------------------------------------------------------------
extern "C" void kernel_launch(void* const* d_in, const int* in_sizes, int n_in,
                              void* d_out, int out_size, void* d_ws, size_t ws_size,
                              hipStream_t stream) {
    const float* x      = (const float*)d_in[0];
    const float* mask   = (const float*)d_in[1];
    const float* Wq     = (const float*)d_in[2];
    const float* bq     = (const float*)d_in[3];
    const float* Wk     = (const float*)d_in[4];
    const float* bk     = (const float*)d_in[5];
    const float* Wv     = (const float*)d_in[6];
    const float* bv     = (const float*)d_in[7];
    const float* alpha1 = (const float*)d_in[8];
    const float* alpha2 = (const float*)d_in[9];
    const float* mq_w1  = (const float*)d_in[10];
    const float* mq_b1  = (const float*)d_in[11];
    const float* mq_w2  = (const float*)d_in[12];
    const float* mq_b2  = (const float*)d_in[13];
    const float* mk_w1  = (const float*)d_in[14];
    const float* mk_b1  = (const float*)d_in[15];
    const float* mk_w2  = (const float*)d_in[16];
    const float* mk_b2  = (const float*)d_in[17];
    const float* ca_w1  = (const float*)d_in[18];
    const float* ca_b1  = (const float*)d_in[19];
    const float* ca_w2  = (const float*)d_in[20];
    const float* ca_b2  = (const float*)d_in[21];
    float* out = (float*)d_out;

    char* ws = (char*)d_ws;
    _Float16* xh    = (_Float16*)(ws);                       // 8 MB [B][HW][64]
    _Float16* maskh = (_Float16*)(ws + (8u  << 20));         // 8 MB [B][HW][64]
    _Float16* uh    = maskh;                                 // reuse after K1
    _Float16* th    = (_Float16*)(ws + (16u << 20));         // 16 MB [2*B][HW][64]
    float*    ao    = (float*)   (ws + (16u << 20));         // 4 MB, reuse after K2
    _Float16* pq    = (_Float16*)(ws + (32u << 20));         // 1 MB [B][N][32]
    _Float16* pk    = pq + (size_t)4 * NN * 32;              // 1 MB
    _Float16* pv    = pk + (size_t)4 * NN * 32;              // 2 MB [B][64][N]
    _Float16* wtap  = (_Float16*)(ws + (36u << 20));         // 144 KB
    _Float16* cw1h  = wtap + 2 * 9 * 64 * 64;
    _Float16* cw2h  = cw1h + 4096;

    pack_data   <<<256, 256, 0, stream>>>(x, mask, xh, maskh);
    pack_weights<<<320, 256, 0, stream>>>(mq_w1, mk_w1, ca_w1, ca_w2, wtap, cw1h, cw2h);
    conv3_wmma  <<<1024, 256, 0, stream>>>(maskh, wtap, mq_b1, mk_b1, th);
    qkv_prep    <<<64, 256, 0, stream>>>(xh, th, Wq, bq, Wk, bk, Wv, bv,
                                         mq_w2, mq_b2, mk_w2, mk_b2,
                                         alpha1, alpha2, pq, pk, pv);
    flash_attn  <<<dim3(64, 4), 128, 0, stream>>>(pq, pk, pv, ao);
    upsample_bilin<<<256, 256, 0, stream>>>(ao, uh);
    final_conv  <<<512, 256, 0, stream>>>(uh, cw1h, cw2h, ca_b1, ca_b2, x, out);
}
